// DCT2_30554397343981
// MI455X (gfx1250) — matrix operand order repaired and checked by reading, not compiled, not run on hardware
//
#include <hip/hip_runtime.h>

// 2-D DCT-II (4096x4096) as  out = C * X * C^T  using split-bf16 (bf16x3)
// WMMA GEMMs on gfx1250.  C[k][n] = cos(pi*(2n+1)*k/8192).
//
// Workspace layout (d_ws), each plane = 4096*4096 bf16 = 32 MB:
//   [0] C_hi  [1] C_lo  [2] X_hi  [3] X_lo  [4] Tt_hi  [5] Tt_lo   (192 MB total)

#define SZ 4096

typedef __attribute__((ext_vector_type(16))) __bf16 v16bf;
typedef __attribute__((ext_vector_type(8)))  __bf16 v8bf;
typedef __attribute__((ext_vector_type(8)))  float  v8f;

static __device__ inline v16bf bf_cat(v8bf a, v8bf b) {
  v16bf r;
#pragma unroll
  for (int i = 0; i < 8; ++i) { r[i] = a[i]; r[i + 8] = b[i]; }
  return r;
}

// ---------------------------------------------------------------------------
// Build the DCT-II cosine matrix, split into bf16 hi/lo.
// Angle reduction is exact: (2n+1)*k < 2^25 fits int32 exactly; period of
// cos(pi*t/8192) is t mod 16384.
// ---------------------------------------------------------------------------
__global__ __launch_bounds__(256) void dct_gen_c(__bf16* __restrict__ Ch,
                                                 __bf16* __restrict__ Cl) {
  size_t idx = (size_t)blockIdx.x * 256 + threadIdx.x;
  int n = (int)(idx & (SZ - 1));
  int k = (int)(idx >> 12);
  int t = ((2 * n + 1) * k) & (4 * SZ - 1);           // exact mod 16384
  float c = cosf((float)t * 3.8349519697141029e-04f); // pi/8192
  __bf16 h = (__bf16)c;
  Ch[idx] = h;
  Cl[idx] = (__bf16)(c - (float)h);
}

// Split fp32 input into bf16 hi/lo planes.
__global__ __launch_bounds__(256) void dct_split_f32(const float* __restrict__ x,
                                                     __bf16* __restrict__ xh,
                                                     __bf16* __restrict__ xl) {
  size_t idx = (size_t)blockIdx.x * 256 + threadIdx.x;
  float v = x[idx];
  __bf16 h = (__bf16)v;
  xh[idx] = h;
  xl[idx] = (__bf16)(v - (float)h);
}

// ---------------------------------------------------------------------------
// Fragment loaders (wave32, 16-bit operands); contiguous k = 16h + s, same for A and B.
// A (16x32, MxK): lane L -> row M=L&15; K chunk {kb..kb+15},
//                 kb = (L>>4)*16.    (two global_load_b128)
// B (32x16, KxN) loaded from transpose-storage BT[n][k] row-major:
//                 lane L -> col N=L&15; K chunk {kb..kb+15}, kb = (L>>4)*16.
// ---------------------------------------------------------------------------
static __device__ inline v16bf load_fragA(const __bf16* __restrict__ p,
                                          int row, int k0, int lane) {
  int m  = lane & 15;
  int kb = (lane >> 4) * 16;
  const __bf16* q = p + (size_t)(row + m) * SZ + (k0 + kb);
  v8bf c0 = *(const v8bf*)(q);
  v8bf c1 = *(const v8bf*)(q + 8);
  return bf_cat(c0, c1);
}

static __device__ inline v16bf load_fragB(const __bf16* __restrict__ p,
                                          int col, int k0, int lane) {
  int n  = lane & 15;
  int kb = (lane >> 4) * 16;
  const __bf16* q = p + (size_t)(col + n) * SZ + (k0 + kb);
  v8bf c0 = *(const v8bf*)(q);
  v8bf c1 = *(const v8bf*)(q + 8);
  return bf_cat(c0, c1);
}

// ---------------------------------------------------------------------------
// NT GEMM:  D[m, c] = sum_k Astore[m][k] * Bstore[c][k]   (both row-major,
// inner dim contiguous).  Split-bf16: D = Ah*Bh + Ah*Bl + Al*Bh, fp32 acc.
// 128 threads = 4 waves (2x2); each wave owns a 32x64 tile (2x4 WMMA tiles);
// block tile 64x128.  Grid divides exactly -> EXEC all ones (WMMA needs it).
// 24 WMMAs per K=32 step vs 24 b128 loads; 8 independent accumulator chains.
//
// store_split_t == 1 : store D transposed, re-split into bf16 hi/lo (feeds
//                      the second GEMM as a row-major NT operand).
// store_split_t == 0 : store fp32 D normally (final output).
// ---------------------------------------------------------------------------
__global__ __launch_bounds__(128) void dct_gemm_nt(
    const __bf16* __restrict__ Ah, const __bf16* __restrict__ Al,
    const __bf16* __restrict__ Bh, const __bf16* __restrict__ Bl,
    float* __restrict__ D, __bf16* __restrict__ Dh, __bf16* __restrict__ Dl,
    int store_split_t) {
  const int lane = threadIdx.x & 31;
  const int wave = threadIdx.x >> 5;
  const int wm = blockIdx.y * 64  + (wave >> 1) * 32;  // 2 tile-rows (M)
  const int wn = blockIdx.x * 128 + (wave & 1) * 64;   // 4 tile-cols (N)

  const v8f vzero = {0.f, 0.f, 0.f, 0.f, 0.f, 0.f, 0.f, 0.f};
  v8f acc[2][4];
#pragma unroll
  for (int i = 0; i < 2; ++i)
#pragma unroll
    for (int j = 0; j < 4; ++j) acc[i][j] = vzero;

  for (int k0 = 0; k0 < SZ; k0 += 32) {
    // Pull next K-chunk cachelines into cache (global_prefetch_b8).
    if (k0 + 64 < SZ) {
      __builtin_prefetch(Ah + (size_t)(wm + lane) * SZ + k0 + 64, 0, 0);
      __builtin_prefetch(Al + (size_t)(wm + lane) * SZ + k0 + 64, 0, 0);
      __builtin_prefetch(Bh + (size_t)(wn + lane) * SZ + k0 + 64, 0, 0);
      __builtin_prefetch(Bl + (size_t)(wn + lane) * SZ + k0 + 64, 0, 0);
      __builtin_prefetch(Bh + (size_t)(wn + 32 + lane) * SZ + k0 + 64, 0, 0);
      __builtin_prefetch(Bl + (size_t)(wn + 32 + lane) * SZ + k0 + 64, 0, 0);
    }

    v16bf ah[2], al[2], bh[4], bl[4];
#pragma unroll
    for (int i = 0; i < 2; ++i) {
      ah[i] = load_fragA(Ah, wm + i * 16, k0, lane);
      al[i] = load_fragA(Al, wm + i * 16, k0, lane);
    }
#pragma unroll
    for (int j = 0; j < 4; ++j) {
      bh[j] = load_fragB(Bh, wn + j * 16, k0, lane);
      bl[j] = load_fragB(Bl, wn + j * 16, k0, lane);
    }

#pragma unroll
    for (int i = 0; i < 2; ++i)
#pragma unroll
      for (int j = 0; j < 4; ++j) {
        acc[i][j] = __builtin_amdgcn_wmma_f32_16x16x32_bf16(
            false, ah[i], false, bh[j], (short)0, acc[i][j], false, false);
        acc[i][j] = __builtin_amdgcn_wmma_f32_16x16x32_bf16(
            false, ah[i], false, bl[j], (short)0, acc[i][j], false, false);
        acc[i][j] = __builtin_amdgcn_wmma_f32_16x16x32_bf16(
            false, al[i], false, bh[j], (short)0, acc[i][j], false, false);
      }
  }

  // C/D layout: VGPR r, lanes 0-15 -> (M=r, N=lane); lanes 16-31 -> (M=r+8).
  const int n  = lane & 15;
  const int mb = (lane >> 4) * 8;

  if (store_split_t) {
    // Transposed store: Dt[n_global][m_global..+7] -- contiguous per lane.
#pragma unroll
    for (int i = 0; i < 2; ++i)
#pragma unroll
      for (int j = 0; j < 4; ++j) {
        int gm = wm + i * 16 + mb;
        int gn = wn + j * 16 + n;
        v8bf h8, l8;
#pragma unroll
        for (int r = 0; r < 8; ++r) {
          float v = acc[i][j][r];
          __bf16 h = (__bf16)v;
          h8[r] = h;
          l8[r] = (__bf16)(v - (float)h);
        }
        *(v8bf*)(Dh + (size_t)gn * SZ + gm) = h8;
        *(v8bf*)(Dl + (size_t)gn * SZ + gm) = l8;
      }
  } else {
#pragma unroll
    for (int i = 0; i < 2; ++i)
#pragma unroll
      for (int j = 0; j < 4; ++j) {
        int gm = wm + i * 16 + mb;
        int gn = wn + j * 16 + n;
#pragma unroll
        for (int r = 0; r < 8; ++r)
          D[(size_t)(gm + r) * SZ + gn] = acc[i][j][r];
      }
  }
}

// ---------------------------------------------------------------------------
extern "C" void kernel_launch(void* const* d_in, const int* in_sizes, int n_in,
                              void* d_out, int out_size, void* d_ws, size_t ws_size,
                              hipStream_t stream) {
  (void)in_sizes; (void)n_in; (void)out_size; (void)ws_size;
  const float* x = (const float*)d_in[0];   // expkM/expkN (d_in[1..2]) unused:
                                            // we build the cosine matrix directly.
  float* out = (float*)d_out;

  const size_t E = (size_t)SZ * SZ;
  __bf16* Ch = (__bf16*)d_ws;
  __bf16* Cl = Ch + E;
  __bf16* Xh = Cl + E;
  __bf16* Xl = Xh + E;
  __bf16* Th = Xl + E;
  __bf16* Tl = Th + E;

  dct_gen_c    <<<(unsigned)(E / 256), 256, 0, stream>>>(Ch, Cl);
  dct_split_f32<<<(unsigned)(E / 256), 256, 0, stream>>>(x, Xh, Xl);

  dim3 gg(SZ / 128, SZ / 64), bb(128);
  // GEMM1: T = X * C^T, stored transposed + re-split -> Tt[kN][m] (bf16 hi/lo)
  dct_gemm_nt<<<gg, bb, 0, stream>>>(Xh, Xl, Ch, Cl, nullptr, Th, Tl, 1);
  // GEMM2: out[k2][kN] = sum_m C[k2][m] * Tt[kN][m]  (fp32 store)
  dct_gemm_nt<<<gg, bb, 0, stream>>>(Ch, Cl, Th, Tl, out, nullptr, nullptr, 0);
}
